// BiMambaModule_49941879718537
// MI455X (gfx1250) — compile-verified, matrix operand order checked by reading, not run on hardware
//
#include <hip/hip_runtime.h>
#include <hip/hip_bf16.h>
#include <math.h>

typedef __attribute__((ext_vector_type(16))) __bf16 v16bf;
typedef __attribute__((ext_vector_type(8)))  __bf16 v8bf;
typedef __attribute__((ext_vector_type(8)))  float  v8f;
typedef __attribute__((ext_vector_type(4)))  unsigned int u32x4;
typedef __attribute__((ext_vector_type(8)))  unsigned int u32x8;

#define BATCH 2
#define LSEQ  4096
#define DMODEL 1024
#define DINNER 2048
#define NSTATE 16
#define NPROJ 96           // dt_rank + 2*d_state
#define DTRANK 64
#define MTOT (BATCH*LSEQ)  // 8192

// build a v16bf fragment from an LDS tile row per the documented 16-bit A/B
// 16x32 layout: lanes 0-15 take K 0..7 & 16..23, lanes 16-31 take 8..15 & 24..31
__device__ __forceinline__ v16bf frag16(const __bf16 (*tile)[40], int row, int hi)
{
    const v8bf lo  = *reinterpret_cast<const v8bf*>(&tile[row][hi * 8]);
    const v8bf hi8 = *reinterpret_cast<const v8bf*>(&tile[row][16 + hi * 8]);
    return __builtin_shufflevector(lo, hi8,
        0,1,2,3,4,5,6,7,8,9,10,11,12,13,14,15);
}

#define WMMA_BF16(a, b, c) \
    __builtin_amdgcn_wmma_f32_16x16x32_bf16(false, (a), false, (b), (short)0, (c), false, false)

// ---------------------------------------------------------------------------
// Generic GEMM: C[M,N] = act( A[M,K] @ W[N,K]^T ), f32 in/out, bf16 WMMA core.
// Block = 256 threads = 8 waves; block tile 64x256; each wave a 32x64 tile
// (2x4 of 16x16 -> 8 WMMAs per K-step; 12 fragment loads -> 2.7x reuse).
// mode 0: none   mode 1: softplus(x + bias[n])   mode 2: x * mask[row]
// ---------------------------------------------------------------------------
__global__ __launch_bounds__(256)
void gemm_wmma_bf16(const float* __restrict__ A, int lda,
                    const float* __restrict__ W, int ldw,
                    float* __restrict__ C, int ldc,
                    int M, int N, int K,
                    const float* __restrict__ bias,
                    const unsigned char* __restrict__ mask,
                    int mode)
{
    // +8 half padding: row stride 80B (16B-aligned), spreads banks
    __shared__ __bf16 lA[64][40];
    __shared__ __bf16 lB[256][40];

    const int tid  = threadIdx.x;
    const int lane = tid & 31;
    const int wave = tid >> 5;
    const int wr   = wave >> 2;     // 0..1 : 32-row wave tile
    const int wc   = wave & 3;      // 0..3 : 64-col wave tile
    const int m16  = lane & 15;
    const int hi   = lane >> 4;

    const int blockRow = blockIdx.y * 64;
    const int blockCol = blockIdx.x * 256;

    // cooperative stage indices
    const int aRow = tid >> 2;          // 0..63
    const int aCol = (tid & 3) * 8;     // 0,8,16,24
    const int nG   = blockCol + tid;    // B: one n-row per thread

    v8f acc[2][4] = {};

    for (int k0 = 0; k0 < K; k0 += 32) {
        // ---- stage A tile (64 x 32) as bf16: 8 floats/thread ----
        {
            const float* ap = A + (size_t)(blockRow + aRow) * lda + (k0 + aCol);
            const float4 a0 = *reinterpret_cast<const float4*>(ap);
            const float4 a1 = *reinterpret_cast<const float4*>(ap + 4);
            lA[aRow][aCol+0] = (__bf16)a0.x;  lA[aRow][aCol+1] = (__bf16)a0.y;
            lA[aRow][aCol+2] = (__bf16)a0.z;  lA[aRow][aCol+3] = (__bf16)a0.w;
            lA[aRow][aCol+4] = (__bf16)a1.x;  lA[aRow][aCol+5] = (__bf16)a1.y;
            lA[aRow][aCol+6] = (__bf16)a1.z;  lA[aRow][aCol+7] = (__bf16)a1.w;
        }
        // ---- stage B tile: rows of W (256 x 32) as bf16: 32 floats/thread ----
        if (nG < N) {
            const float* wp = W + (size_t)nG * ldw + k0;
            #pragma unroll
            for (int q = 0; q < 8; ++q) {
                const float4 wv = *reinterpret_cast<const float4*>(wp + q * 4);
                lB[tid][q*4 + 0] = (__bf16)wv.x;
                lB[tid][q*4 + 1] = (__bf16)wv.y;
                lB[tid][q*4 + 2] = (__bf16)wv.z;
                lB[tid][q*4 + 3] = (__bf16)wv.w;
            }
        } else {
            #pragma unroll
            for (int j = 0; j < 32; ++j) lB[tid][j] = (__bf16)0.0f;
        }
        __syncthreads();

        // prefetch next K tile (global_prefetch_b8 on gfx1250)
        if (k0 + 32 < K) {
            __builtin_prefetch(A + (size_t)(blockRow + aRow) * lda + (k0 + 32 + aCol), 0, 1);
            __builtin_prefetch(W + (size_t)(nG < N ? nG : 0) * ldw + (k0 + 32), 0, 1);
        }

        // ---- fragments: 2 A rows, 4 B cols -> 8 WMMAs ----
        const v16bf a0 = frag16(lA, wr * 32      + m16, hi);
        const v16bf a1 = frag16(lA, wr * 32 + 16 + m16, hi);
        v16bf bf[4];
        #pragma unroll
        for (int t = 0; t < 4; ++t)
            bf[t] = frag16(lB, wc * 64 + t * 16 + m16, hi);

        #pragma unroll
        for (int t = 0; t < 4; ++t) {
            acc[0][t] = WMMA_BF16(a0, bf[t], acc[0][t]);
            acc[1][t] = WMMA_BF16(a1, bf[t], acc[1][t]);
        }

        __syncthreads();
    }

    // ---- epilogue: C/D layout — lane col = lane&15, row = v + 8*(lane>>4) ----
    #pragma unroll
    for (int ti = 0; ti < 2; ++ti) {
        #pragma unroll
        for (int tj = 0; tj < 4; ++tj) {
            const int nOut = blockCol + wc * 64 + tj * 16 + m16;
            #pragma unroll
            for (int v = 0; v < 8; ++v) {
                const int mOut = blockRow + wr * 32 + ti * 16 + v + 8 * hi;
                float val = acc[ti][tj][v];
                if (mode == 1) {
                    const float xb = val + bias[nOut < N ? nOut : 0];
                    val = (xb > 20.0f) ? xb : log1pf(__expf(xb));   // softplus
                } else if (mode == 2) {
                    val *= (mask[mOut] ? 1.0f : 0.0f);
                }
                if (nOut < N) C[(size_t)mOut * ldc + nOut] = val;
            }
        }
    }
}

// ---------------------------------------------------------------------------
// Depthwise causal conv1d (k=4) + SiLU.  xi = xz[..., 0:DINNER].
// rev=1 reads the time-flipped sequence (backward branch).
// ---------------------------------------------------------------------------
__global__ __launch_bounds__(256)
void conv_silu(const float* __restrict__ xz, float* __restrict__ xt,
               const float* __restrict__ cw, const float* __restrict__ cb,
               int rev)
{
    const long long idx = (long long)blockIdx.x * 256 + threadIdx.x;
    const int d = (int)(idx % DINNER);
    const int l = (int)((idx / DINNER) % LSEQ);
    const int b = (int)(idx / ((long long)DINNER * LSEQ));

    float acc = cb[d];
    #pragma unroll
    for (int j = 0; j < 4; ++j) {
        const int t = l - 3 + j;
        if (t >= 0) {
            const int src = rev ? (LSEQ - 1 - t) : t;
            acc += cw[d * 4 + j] * xz[((size_t)(b * LSEQ + src)) * (2 * DINNER) + d];
        }
    }
    xt[idx] = acc / (1.0f + __expf(-acc));   // silu
}

// ---------------------------------------------------------------------------
// Selective scan. One thread per (b,d); 16 states in registers; serial in L.
// B_t/C_t (shared across d) staged into LDS per 64-step chunk either by
//  - TDM: one tensor_load_to_lds (D# built per ISA 08 §8.3-8.6; TENSORcnt), or
//  - per-lane GLOBAL_LOAD_ASYNC_TO_LDS_B128 (ASYNCcnt).
// Fuses +u*D and *silu(z); writes y in place over u (xt buffer).
// ---------------------------------------------------------------------------
__global__ __launch_bounds__(256)
void scan_kernel(float* __restrict__ u_y,            // xt in, y out (in place)
                 const float* __restrict__ dt,
                 const float* __restrict__ proj,     // [M, 96]; B at 64, C at 80
                 const float* __restrict__ xz,       // z = xz[..., DINNER:]
                 const float* __restrict__ A_log,
                 const float* __restrict__ Dp,
                 int rev, int use_tdm)
{
    __shared__ float sBC[64][32];   // [step][0:16)=B_t, [16:32)=C_t

    const int tid = threadIdx.x;
    const int b   = blockIdx.x / (DINNER / 256);
    const int d   = (blockIdx.x % (DINNER / 256)) * 256 + tid;

    float Areg[NSTATE], h[NSTATE];
    #pragma unroll
    for (int n = 0; n < NSTATE; ++n) {
        Areg[n] = -__expf(A_log[d * NSTATE + n]);
        h[n] = 0.0f;
    }
    const float Dd = Dp[d];

    // low 32 bits of a generic pointer to __shared__ == LDS byte offset
    const unsigned ldsBase = (unsigned)(size_t)(&sBC[0][0]);

    for (int l0 = 0; l0 < LSEQ; l0 += 64) {
        if (use_tdm) {
            // ---- Tensor Data Mover: 2D tile 32(f32) x 64 rows, stride 96 f32 ----
            if ((tid >> 5) == 0) {   // one wave issues the DMA (EXEC ignored by TDM)
                const unsigned long long ga = (unsigned long long)(size_t)
                    (proj + ((size_t)(b * LSEQ + l0)) * NPROJ + DTRANK);
                u32x4 g0;
                g0[0] = 1u;                                   // count=1, user mode
                g0[1] = ldsBase;                              // lds_addr
                g0[2] = (unsigned)(ga & 0xffffffffull);       // global_addr[31:0]
                g0[3] = (unsigned)((ga >> 32) & 0x1ffffffu)   // global_addr[56:32]
                        | (2u << 30);                         // type=2 (image)
                u32x8 g1;
                g1[0] = 2u << 16;      // data_size=4B; no barrier/iterate/pad
                g1[1] = 32u << 16;     // tensor_dim0 = 32
                g1[2] = 64u << 16;     // tensor_dim1 = 64
                g1[3] = 32u << 16;     // tile_dim0 = 32
                g1[4] = 64u;           // tile_dim1 = 64 (tile_dim2 = 0)
                g1[5] = 96u;           // tensor_dim0_stride = 96
                g1[6] = 0u;            // (dim1_stride unused for 2D)
                g1[7] = 0u;
                const u32x4 gz = {0u, 0u, 0u, 0u};            // groups 2/3: zeros -> 2D
                asm volatile("tensor_load_to_lds %0, %1, %2, %3"
                             :: "s"(g0), "s"(g1), "s"(gz), "s"(gz) : "memory");
                __builtin_amdgcn_s_wait_tensorcnt(0);
            }
            __syncthreads();
        } else {
            // ---- async DMA of 64x32 f32 chunk (8 KB): 2 x b128 per lane ----
            #pragma unroll
            for (int j = 0; j < 2; ++j) {
                const int c   = tid * 2 + j;        // 0..511 : 16B chunk index
                const int row = c >> 3;             // 0..63
                const int seg = c & 7;              // 0..7 : 4 floats each
                const unsigned lds = ldsBase + (unsigned)((row * 32 + seg * 4) * 4);
                const unsigned long long ga = (unsigned long long)(size_t)
                    (proj + ((size_t)(b * LSEQ + l0 + row)) * NPROJ + DTRANK + seg * 4);
                asm volatile("global_load_async_to_lds_b128 %0, %1, off"
                             :: "v"(lds), "v"(ga) : "memory");
            }
            asm volatile("s_wait_asynccnt 0x0" ::: "memory");
            __syncthreads();
        }

        for (int ll = 0; ll < 64; ++ll) {
            const int l = l0 + ll;
            const size_t base = ((size_t)(b * LSEQ + l)) * DINNER + d;
            const float dtv = dt[base];
            const float uv  = u_y[base];
            const float du  = dtv * uv;
            float y = 0.0f;
            #pragma unroll
            for (int n = 0; n < NSTATE; ++n) {
                const float dA = __expf(dtv * Areg[n]);
                h[n] = dA * h[n] + du * sBC[ll][n];
                y += h[n] * sBC[ll][NSTATE + n];
            }
            y += uv * Dd;
            const int zl = rev ? (LSEQ - 1 - l) : l;
            const float zv = xz[((size_t)(b * LSEQ + zl)) * (2 * DINNER) + DINNER + d];
            y *= zv / (1.0f + __expf(-zv));     // * silu(z)
            u_y[base] = y;
        }
        __syncthreads();
    }
}

// y_f[b,l,d] += y_b[b, L-1-l, d]
__global__ __launch_bounds__(256)
void combine_kernel(float* __restrict__ yf, const float* __restrict__ yb)
{
    const long long idx = (long long)blockIdx.x * 256 + threadIdx.x;
    const int d = (int)(idx % DINNER);
    const int l = (int)((idx / DINNER) % LSEQ);
    const int b = (int)(idx / ((long long)DINNER * LSEQ));
    yf[idx] += yb[((size_t)(b * LSEQ + (LSEQ - 1 - l))) * DINNER + d];
}

// second tuple output: pass masks through
__global__ __launch_bounds__(256)
void write_masks(const unsigned char* __restrict__ m, float* __restrict__ out)
{
    const int i = blockIdx.x * 256 + threadIdx.x;
    if (i < BATCH * LSEQ) out[i] = m[i] ? 1.0f : 0.0f;
}

// ---------------------------------------------------------------------------
extern "C" void kernel_launch(void* const* d_in, const int* in_sizes, int n_in,
                              void* d_out, int out_size, void* d_ws, size_t ws_size,
                              hipStream_t stream) {
    const float* x           = (const float*)d_in[0];
    const unsigned char* msk = (const unsigned char*)d_in[1];
    const float* in_proj_w   = (const float*)d_in[2];
    const float* conv_w      = (const float*)d_in[3];
    const float* conv_b      = (const float*)d_in[4];
    const float* x_proj_w    = (const float*)d_in[5];
    const float* dt_proj_w   = (const float*)d_in[6];
    const float* dt_proj_b   = (const float*)d_in[7];
    const float* A_log       = (const float*)d_in[8];
    const float* Dp          = (const float*)d_in[9];
    const float* conv_w_b    = (const float*)d_in[10];
    const float* conv_b_b    = (const float*)d_in[11];
    const float* x_proj_w_b  = (const float*)d_in[12];
    const float* dt_proj_w_b = (const float*)d_in[13];
    const float* dt_proj_b_b = (const float*)d_in[14];
    const float* A_b_log     = (const float*)d_in[15];
    const float* D_b         = (const float*)d_in[16];
    const float* out_proj_w  = (const float*)d_in[17];
    float* out = (float*)d_out;
    float* ws  = (float*)d_ws;

    // workspace carve-up (floats)
    float* xz    = ws;                       // 8192*4096
    float* xtf   = xz    + 33554432ull;      // 8192*2048
    float* xtb   = xtf   + 16777216ull;      // 8192*2048
    float* projf = xtb   + 16777216ull;      // 8192*96
    float* projb = projf + 786432ull;
    float* dtf   = projb + 786432ull;        // 8192*2048
    float* dtb   = dtf   + 16777216ull;

    const dim3 blk(256);
    const int elemBlocks = (BATCH * LSEQ * DINNER) / 256;   // 65536

    // 1) in_proj: xz = x @ in_proj_w^T   [8192,4096]
    gemm_wmma_bf16<<<dim3((2*DINNER)/256, MTOT/64), blk, 0, stream>>>(
        x, DMODEL, in_proj_w, DMODEL, xz, 2*DINNER,
        MTOT, 2*DINNER, DMODEL, nullptr, nullptr, 0);

    // 2) causal conv + silu (fwd & time-flipped bwd)
    conv_silu<<<elemBlocks, blk, 0, stream>>>(xz, xtf, conv_w,   conv_b,   0);
    conv_silu<<<elemBlocks, blk, 0, stream>>>(xz, xtb, conv_w_b, conv_b_b, 1);

    // 3) x_proj: proj = xt @ x_proj_w^T  [8192,96]
    gemm_wmma_bf16<<<dim3(1, MTOT/64), blk, 0, stream>>>(
        xtf, DINNER, x_proj_w,   DINNER, projf, NPROJ, MTOT, NPROJ, DINNER,
        nullptr, nullptr, 0);
    gemm_wmma_bf16<<<dim3(1, MTOT/64), blk, 0, stream>>>(
        xtb, DINNER, x_proj_w_b, DINNER, projb, NPROJ, MTOT, NPROJ, DINNER,
        nullptr, nullptr, 0);

    // 4) dt = softplus(dt_slice @ dt_proj_w^T + b)  [8192,2048]; dt_slice = proj[:, :64]
    gemm_wmma_bf16<<<dim3(DINNER/256, MTOT/64), blk, 0, stream>>>(
        projf, NPROJ, dt_proj_w,   DTRANK, dtf, DINNER, MTOT, DINNER, DTRANK,
        dt_proj_b,   nullptr, 1);
    gemm_wmma_bf16<<<dim3(DINNER/256, MTOT/64), blk, 0, stream>>>(
        projb, NPROJ, dt_proj_w_b, DTRANK, dtb, DINNER, MTOT, DINNER, DTRANK,
        dt_proj_b_b, nullptr, 1);

    // 5) selective scans (fwd uses TDM staging, bwd uses async-to-LDS staging)
    scan_kernel<<<BATCH * (DINNER/256), blk, 0, stream>>>(
        xtf, dtf, projf, xz, A_log,   Dp,  0, 1);
    scan_kernel<<<BATCH * (DINNER/256), blk, 0, stream>>>(
        xtb, dtb, projb, xz, A_b_log, D_b, 1, 0);

    // 6) y = y_f + flip(y_b)
    combine_kernel<<<elemBlocks, blk, 0, stream>>>(xtf, xtb);

    // 7) out = mask * (y @ out_proj_w^T)   [8192,1024]
    gemm_wmma_bf16<<<dim3(DMODEL/256, MTOT/64), blk, 0, stream>>>(
        xtf, DINNER, out_proj_w, DINNER, out, DMODEL,
        MTOT, DMODEL, DINNER, nullptr, msk, 2);

    // 8) tuple output #2: masks
    write_masks<<<(BATCH*LSEQ + 255)/256, blk, 0, stream>>>(
        msk, out + (size_t)MTOT * DMODEL);
}
